// CachePolicyModel_13460427505855
// MI455X (gfx1250) — matrix operand order repaired and checked by reading, not compiled, not run on hardware
//
#include <hip/hip_runtime.h>
#include <hip/hip_bf16.h>

// ---------------- CDNA5 WMMA types ----------------
typedef __attribute__((ext_vector_type(16))) __bf16 v16bf;
typedef __attribute__((ext_vector_type(8)))  float  v8f;

#define B_   64
#define L_   1024
#define H_   256
#define T_   128
#define DID  128
#define DSZ  64

// A-matrix fragment (16x32 bf16, ISA 7.12.2): lane holds row M=lane%16;
// lanes 0-15 cover K = {kbase..kbase+7, kbase+16..kbase+23},
// lanes 16-31 cover K = {kbase+8..kbase+15, kbase+24..kbase+31}.
__device__ inline v16bf fragA(const __bf16* base, int ld, int row0, int kbase, int lane) {
  const __bf16* p = base + (row0 + (lane & 15)) * ld + kbase + ((lane >> 4) ? 8 : 0);
  v16bf a;
#pragma unroll
  for (int j = 0; j < 8; ++j) a[j] = p[j];
#pragma unroll
  for (int j = 0; j < 8; ++j) a[8 + j] = p[16 + j];
  return a;
}

// B-matrix fragment (32x16 bf16): lane holds column N=lane%16; lanes 0-15
// hold K = {kbase..kbase+7, kbase+16..kbase+23}, lanes 16-31 hold
// K = {kbase+8..kbase+15, kbase+24..kbase+31} (same as A).
// Storage is row-per-N with K contiguous (i.e. B^T row-major), so 16 reads.
__device__ inline v16bf fragB(const __bf16* base, int ld, int n0, int kbase, int lane) {
  const __bf16* p = base + (n0 + (lane & 15)) * ld + kbase + ((lane >> 4) ? 8 : 0);
  v16bf b;
#pragma unroll
  for (int j = 0; j < 8; ++j) b[j] = p[j];
#pragma unroll
  for (int j = 0; j < 8; ++j) b[8 + j] = p[16 + j];
  return b;
}

__device__ inline float sigm(float x) { return 1.f / (1.f + __expf(-x)); }

// ---------------- Kernel 0: prep (attn_W -> bf16, pos-head dots) ----------------
__global__ void prep_kernel(const float* __restrict__ attn_W,
                            const float* __restrict__ pos_table,
                            const float* __restrict__ scorer_W,
                            const float* __restrict__ reuse_W,
                            __bf16* __restrict__ attnbf,
                            float* __restrict__ ps, float* __restrict__ pr) {
  int tid = threadIdx.x;
  for (int i = tid; i < DID * H_; i += 256) attnbf[i] = (__bf16)attn_W[i];
  if (tid < T_) {
    float a = 0.f, r = 0.f;
    for (int p = 0; p < 128; ++p) {
      float pv = pos_table[tid * 128 + p];
      a += pv * scorer_W[H_ + p];
      r += pv * reuse_W[H_ + p];
    }
    ps[tid] = a; pr[tid] = r;
  }
}

// ---------------- Kernel 1: LSTM cell + bf16 history ----------------
__global__ void lstm_kernel(const int* __restrict__ obj_id, const int* __restrict__ obj_size,
                            const float* __restrict__ c0, const float* __restrict__ h0,
                            const float* __restrict__ history,
                            const float* __restrict__ obj_id_table,
                            const float* __restrict__ obj_size_table,
                            const float* __restrict__ W_ih, const float* __restrict__ W_hh,
                            const float* __restrict__ b_ih, const float* __restrict__ b_hh,
                            __bf16* __restrict__ histbf) {
  __shared__ float xs[DID + DSZ];
  __shared__ float hs[H_];
  int b = blockIdx.x, j = threadIdx.x;
  int id = obj_id[b], sz = obj_size[b];
  if (j < DID) xs[j] = obj_id_table[(size_t)id * DID + j];
  else if (j < DID + DSZ) xs[j] = obj_size_table[sz * DSZ + (j - DID)];
  hs[j] = h0[b * H_ + j];
  __syncthreads();
  float g4[4];
#pragma unroll
  for (int gi = 0; gi < 4; ++gi) {
    int row = gi * H_ + j;
    const float* wi = W_ih + (size_t)row * (DID + DSZ);
    const float* wh = W_hh + (size_t)row * H_;
    float acc = b_ih[row] + b_hh[row];
    for (int k = 0; k < DID + DSZ; ++k) acc += xs[k] * wi[k];
    for (int k = 0; k < H_; ++k) acc += hs[k] * wh[k];
    g4[gi] = acc;
  }
  float c = sigm(g4[1]) * c0[b * H_ + j] + sigm(g4[0]) * tanhf(g4[2]);
  float h = sigm(g4[3]) * tanhf(c);
  histbf[((size_t)b * T_ + (T_ - 1)) * H_ + j] = (__bf16)h;
  for (int t = 0; t < T_ - 1; ++t)
    histbf[((size_t)b * T_ + t) * H_ + j] = (__bf16)history[((size_t)b * (T_ - 1) + t) * H_ + j];
}

// ---------------- Kernel 2: Mt[b,t,d] = hist[b,t,:]·attn_W[d,:]  (WMMA) + vs/vr ----------------
__global__ void mt_kernel(const __bf16* __restrict__ histbf, const __bf16* __restrict__ attnbf,
                          const float* __restrict__ scorer_W, const float* __restrict__ reuse_W,
                          const float* __restrict__ ps, const float* __restrict__ pr,
                          __bf16* __restrict__ Mtbf, float* __restrict__ vs, float* __restrict__ vr) {
  __shared__ __bf16 hl[64 * 264];  // 64 t-rows x 256 h (pad 8)
  int b = blockIdx.x >> 1, t0 = (blockIdx.x & 1) * 64;
  int tid = threadIdx.x;
  for (int i = tid; i < 64 * H_; i += 256) {
    int r = i >> 8, cc = i & 255;
    hl[r * 264 + cc] = histbf[((size_t)b * T_ + t0 + r) * H_ + cc];
  }
  __syncthreads();
  int w = tid >> 5, lane = tid & 31;
  for (int ti = 0; ti < 4; ++ti) {
    int tile = w + ti * 8;
    int tt = tile >> 3, dd = tile & 7;  // 4 t-tiles x 8 d-tiles
    v8f acc = {};
#pragma unroll
    for (int ks = 0; ks < 8; ++ks) {  // K = 256
      v16bf a  = fragA(hl, 264, tt * 16, ks * 32, lane);
      v16bf bm = fragB(attnbf, H_, dd * 16, ks * 32, lane);  // attn_W[d,h] row-major: h contiguous
      acc = __builtin_amdgcn_wmma_f32_16x16x32_bf16(false, a, false, bm, (short)0, acc, false, false);
    }
    int tg = t0 + tt * 16 + ((lane >> 4) ? 8 : 0);
    int d  = dd * 16 + (lane & 15);
#pragma unroll
    for (int i = 0; i < 8; ++i)
      Mtbf[((size_t)b * T_ + tg + i) * DID + d] = (__bf16)acc[i];
  }
  __syncthreads();
  int sel = tid >> 6, tl = tid & 63;
  if (sel < 2) {
    const float* wv = sel ? reuse_W : scorer_W;
    float acc = 0.f;
    for (int h = 0; h < H_; ++h) acc += (float)hl[tl * 264 + h] * wv[h];
    int t = t0 + tl;
    if (sel) vr[b * T_ + t] = acc + pr[t];
    else     vs[b * T_ + t] = acc + ps[t];
  }
}

// ---------------- Kernel 3: S = Q @ Mt^T (WMMA), fused softmax + heads ----------------
__global__ void attn_kernel(const int* __restrict__ cache_lines,
                            const float* __restrict__ obj_id_table,
                            const __bf16* __restrict__ Mtbf,
                            const float* __restrict__ vs, const float* __restrict__ vr,
                            const float* __restrict__ scorer_b, const float* __restrict__ reuse_b,
                            float* __restrict__ zline, float* __restrict__ out_reuse) {
  __shared__ __bf16 Qs[32 * 136];    // 32 l-rows x 128 d
  __shared__ __bf16 Ms[128 * 136];   // 128 t-rows x 128 d  (== B^T storage for S)
  __shared__ float  Ss[32 * 132];    // scores tile
  __shared__ float  vsr[2][T_];
  __shared__ float  red[32][8][3];
  __shared__ int    ids[32];
  int b = blockIdx.x >> 5, l0 = (blockIdx.x & 31) * 32;
  int tid = threadIdx.x;
  if (tid < 32) ids[tid] = cache_lines[b * L_ + l0 + tid];
  if (tid < T_) vsr[0][tid] = vs[b * T_ + tid];
  else          vsr[1][tid - T_] = vr[b * T_ + (tid - T_)];
  __syncthreads();
  for (int i = tid; i < 32 * DID; i += 256) {
    int r = i >> 7, cc = i & 127;
    Qs[r * 136 + cc] = (__bf16)obj_id_table[(size_t)ids[r] * DID + cc];
  }
  for (int i = tid; i < T_ * DID; i += 256) {
    int r = i >> 7, cc = i & 127;
    Ms[r * 136 + cc] = Mtbf[((size_t)b * T_ + r) * DID + cc];
  }
  __syncthreads();
  int w = tid >> 5, lane = tid & 31;
  int lb = w & 1, tg = w >> 1;  // 2 l-blocks x 8 t-tiles, 2 tiles/wave
#pragma unroll
  for (int j = 0; j < 2; ++j) {
    int tt = tg + j * 4;
    v8f acc = {};
#pragma unroll
    for (int ks = 0; ks < 4; ++ks) {  // K = 128
      v16bf a  = fragA(Qs, 136, lb * 16, ks * 32, lane);
      v16bf bm = fragB(Ms, 136, tt * 16, ks * 32, lane);
      acc = __builtin_amdgcn_wmma_f32_16x16x32_bf16(false, a, false, bm, (short)0, acc, false, false);
    }
    int rr = lb * 16 + ((lane >> 4) ? 8 : 0);
    int cc = tt * 16 + (lane & 15);
#pragma unroll
    for (int i = 0; i < 8; ++i) Ss[(rr + i) * 132 + cc] = acc[i];
  }
  __syncthreads();
  // row softmax over t (128) fused with vs/vr reductions: 8 threads per row
  int r = tid >> 3, q = tid & 7;
  const float* srow = &Ss[r * 132];
  float m = -1e30f;
  for (int t = q * 16; t < q * 16 + 16; ++t) m = fmaxf(m, srow[t]);
  red[r][q][0] = m;
  __syncthreads();
  float rm = red[r][0][0];
#pragma unroll
  for (int k = 1; k < 8; ++k) rm = fmaxf(rm, red[r][k][0]);
  __syncthreads();
  float es = 0.f, as = 0.f, ar = 0.f;
  for (int t = q * 16; t < q * 16 + 16; ++t) {
    float e = __expf(srow[t] - rm);
    es += e; as += e * vsr[0][t]; ar += e * vsr[1][t];
  }
  red[r][q][0] = es; red[r][q][1] = as; red[r][q][2] = ar;
  __syncthreads();
  if (q == 0) {
    float Z = 0.f, nS = 0.f, nR = 0.f;
#pragma unroll
    for (int k = 0; k < 8; ++k) { Z += red[r][k][0]; nS += red[r][k][1]; nR += red[r][k][2]; }
    int l = l0 + r;
    zline[b * L_ + l]     = nS / Z + scorer_b[0];
    out_reuse[b * L_ + l] = nR / Z + reuse_b[0];
  }
}

// ---------------- Kernel 4: masked softmax over L ----------------
__global__ void renorm_kernel(const float* __restrict__ zline, const int* __restrict__ lengths,
                              float* __restrict__ probs) {
  __shared__ float red[256];
  int b = blockIdx.x, tid = threadIdx.x;
  int valid = lengths[b] > 1 ? lengths[b] : 1;
  float m = -1e30f;
  for (int l = tid; l < L_; l += 256)
    if (l < valid) m = fmaxf(m, zline[b * L_ + l]);
  red[tid] = m; __syncthreads();
  for (int s = 128; s > 0; s >>= 1) { if (tid < s) red[tid] = fmaxf(red[tid], red[tid + s]); __syncthreads(); }
  float rm = red[0]; __syncthreads();
  float es = 0.f;
  for (int l = tid; l < L_; l += 256)
    if (l < valid) es += __expf(zline[b * L_ + l] - rm);
  red[tid] = es; __syncthreads();
  for (int s = 128; s > 0; s >>= 1) { if (tid < s) red[tid] += red[tid + s]; __syncthreads(); }
  float Z = red[0];
  for (int l = tid; l < L_; l += 256)
    probs[b * L_ + l] = (l < valid) ? __expf(zline[b * L_ + l] - rm) / Z : 0.f;
}

// ---------------- Host launcher ----------------
extern "C" void kernel_launch(void* const* d_in, const int* in_sizes, int n_in,
                              void* d_out, int out_size, void* d_ws, size_t ws_size,
                              hipStream_t stream) {
  const int*   obj_id         = (const int*)  d_in[0];
  const int*   obj_size       = (const int*)  d_in[1];
  const int*   cache_lines    = (const int*)  d_in[2];
  const int*   lengths        = (const int*)  d_in[3];
  const float* c0             = (const float*)d_in[4];
  const float* h0             = (const float*)d_in[5];
  const float* history        = (const float*)d_in[6];
  const float* obj_id_table   = (const float*)d_in[7];
  const float* obj_size_table = (const float*)d_in[8];
  const float* pos_table      = (const float*)d_in[9];
  const float* W_ih           = (const float*)d_in[10];
  const float* W_hh           = (const float*)d_in[11];
  const float* b_ih           = (const float*)d_in[12];
  const float* b_hh           = (const float*)d_in[13];
  const float* attn_W         = (const float*)d_in[14];
  const float* scorer_W       = (const float*)d_in[15];
  const float* scorer_b       = (const float*)d_in[16];
  const float* reuse_W        = (const float*)d_in[17];
  const float* reuse_b        = (const float*)d_in[18];

  char* ws = (char*)d_ws;
  size_t off = 0;
  auto take = [&](size_t bytes) { char* p = ws + off; off = (off + bytes + 255) & ~(size_t)255; return p; };
  __bf16* histbf = (__bf16*)take((size_t)B_ * T_ * H_ * 2);
  __bf16* attnbf = (__bf16*)take((size_t)DID * H_ * 2);
  float*  ps     = (float*) take(T_ * 4);
  float*  pr     = (float*) take(T_ * 4);
  __bf16* Mtbf   = (__bf16*)take((size_t)B_ * T_ * DID * 2);
  float*  vs     = (float*) take((size_t)B_ * T_ * 4);
  float*  vr     = (float*) take((size_t)B_ * T_ * 4);
  float*  zline  = (float*) take((size_t)B_ * L_ * 4);

  float* probs     = (float*)d_out;
  float* out_reuse = (float*)d_out + (size_t)B_ * L_;

  prep_kernel<<<1, 256, 0, stream>>>(attn_W, pos_table, scorer_W, reuse_W, attnbf, ps, pr);
  lstm_kernel<<<B_, 256, 0, stream>>>(obj_id, obj_size, c0, h0, history, obj_id_table,
                                      obj_size_table, W_ih, W_hh, b_ih, b_hh, histbf);
  mt_kernel<<<B_ * 2, 256, 0, stream>>>(histbf, attnbf, scorer_W, reuse_W, ps, pr, Mtbf, vs, vr);
  attn_kernel<<<B_ * 32, 256, 0, stream>>>(cache_lines, obj_id_table, Mtbf, vs, vr,
                                           scorer_b, reuse_b, zline, out_reuse);
  renorm_kernel<<<B_, 256, 0, stream>>>(zline, lengths, probs);
}
